// MultiHeadAttention_19576460935456
// MI455X (gfx1250) — matrix operand order repaired and checked by reading, not compiled, not run on hardware
//
#include <hip/hip_runtime.h>

// ---------------------------------------------------------------------------
// MI455X (gfx1250) multi-head attention forward, bf16 WMMA pipeline with
// double-buffered async global->LDS tile streaming (ASYNCcnt path).
// ---------------------------------------------------------------------------

typedef __bf16 bf16;
typedef __attribute__((ext_vector_type(16))) __bf16 v16bf;
typedef __attribute__((ext_vector_type(8)))  float  v8f;
typedef int v4i __attribute__((vector_size(16)));   // matches builtin pointee

#define AS1 __attribute__((address_space(1)))
#define AS3 __attribute__((address_space(3)))

#if __has_builtin(__builtin_amdgcn_global_load_async_to_lds_b128)
#define HAS_ASYNC_LDS 1
#else
#define HAS_ASYNC_LDS 0
#endif

#define D_MODEL 1024
#define SEQ     2048
#define BATCH   2
#define HEADS   16
#define DK      64
#define M_TOT   (BATCH * SEQ)         // 4096 rows

// GEMM tiling
#define TILE_M 128
#define TILE_N 64
#define TILE_K 32
#define LDS_A  (TILE_K + 8)           // 40 (pad vs bank conflicts)
#define LDS_B  (TILE_K + 8)

// Attention tiling
#define ATT_LD 72                     // 64 + 8 pad (keeps 16B-aligned rows)

// 0.125 * log2(e): folds softmax scale + base-2 exp into Q
#define Q_SCALE 0.18033688011112042f

__device__ __forceinline__ v8f wmma_bf16(v16bf a, v16bf b, v8f c) {
  return __builtin_amdgcn_wmma_f32_16x16x32_bf16(
      /*neg_a=*/false, a, /*neg_b=*/false, b,
      /*c_mod=*/(short)0, c, /*reuse_a=*/false, /*reuse_b=*/false);
}

// 16-byte global -> LDS copy: async on CDNA5 (ASYNCcnt), sync fallback.
__device__ __forceinline__ void copy16_g2l(const bf16* g, bf16* l) {
#if HAS_ASYNC_LDS
  __builtin_amdgcn_global_load_async_to_lds_b128(
      (AS1 v4i*)(void*)g, (AS3 v4i*)(void*)l, 0, 0);
#else
  *(uint4*)((void*)l) = *(const uint4*)g;
#endif
}

__device__ __forceinline__ void async_copy_fence() {
#if HAS_ASYNC_LDS
#if __has_builtin(__builtin_amdgcn_s_wait_asynccnt)
  __builtin_amdgcn_s_wait_asynccnt(0);
#else
  asm volatile("s_wait_asynccnt 0" ::: "memory");
#endif
#endif
}

// A fragment (16x32 bf16), row-major source with leading dim `ld`.
// ISA layout: lane&15 = M row; lanes 0-15 hold K 0-7 / 16-23, lanes 16-31 hold
// K 8-15 / 24-31.
__device__ __forceinline__ v16bf frag_a(const bf16* p, int ld, int lane) {
  const bf16* q = p + (lane & 15) * ld + ((lane >> 4) << 3);
  v16bf a;
#pragma unroll
  for (int e = 0; e < 8; ++e)  a[e] = q[e];
#pragma unroll
  for (int e = 8; e < 16; ++e) a[e] = q[e + 8];
  return a;
}

// B fragment (32x16 bf16) from [N][K] row-major source with leading dim `ld`.
// ISA layout: lane&15 = N col; lanes 0-15 hold K 0-7 / 16-23, lanes 16-31 hold
// K 8-15 / 24-31.
__device__ __forceinline__ v16bf frag_b(const bf16* p, int ld, int lane) {
  const bf16* q = p + (lane & 15) * ld + ((lane >> 4) << 3);
  v16bf b;
#pragma unroll
  for (int e = 0; e < 8; ++e)  b[e] = q[e];
#pragma unroll
  for (int e = 8; e < 16; ++e) b[e] = q[e + 8];
  return b;
}

__device__ __forceinline__ v8f v8f_zero() {
  v8f z = {0.f, 0.f, 0.f, 0.f, 0.f, 0.f, 0.f, 0.f};
  return z;
}

// ---------------------------------------------------------------------------
// Kernel 1: fused QKV projection.  y = x @ W^T + b  (fp32 in, bf16 out).
// Q/K laid out [B][H][S][DK]; V laid out TRANSPOSED [B][H][DK][S] so that the
// attention kernel can stream V tiles with pure async b128 copies.
// Q additionally pre-scaled by 1/sqrt(dk)*log2e.
// grid = (M/128, N/64, 3), block = 256 (8 waves, 4x2 wave grid, 2x2 frags/wave)
// ---------------------------------------------------------------------------
__global__ __launch_bounds__(256)
void qkv_proj_kernel(const float* __restrict__ x,
                     const float* __restrict__ Wq, const float* __restrict__ Wk,
                     const float* __restrict__ Wv,
                     const float* __restrict__ bq, const float* __restrict__ bk,
                     const float* __restrict__ bv,
                     bf16* __restrict__ Qo, bf16* __restrict__ Ko,
                     bf16* __restrict__ Vo) {
  __shared__ bf16 As[TILE_M * LDS_A];
  __shared__ bf16 Bs[TILE_N * LDS_B];

  const int z = blockIdx.z;
  const float* W    = (z == 0) ? Wq : (z == 1) ? Wk : Wv;
  const float* bias = (z == 0) ? bq : (z == 1) ? bk : bv;
  bf16*        out  = (z == 0) ? Qo : (z == 1) ? Ko : Vo;
  const float oscale = (z == 0) ? Q_SCALE : 1.0f;

  const int tid  = threadIdx.x;
  const int wave = tid >> 5;
  const int lane = tid & 31;
  const int wm = wave & 3;   // 0..3
  const int wn = wave >> 2;  // 0..1
  const int m0 = blockIdx.x * TILE_M;
  const int n0 = blockIdx.y * TILE_N;

  v8f acc[2][2];
#pragma unroll
  for (int i = 0; i < 2; ++i)
#pragma unroll
    for (int j = 0; j < 2; ++j) acc[i][j] = v8f_zero();

  for (int k0 = 0; k0 < D_MODEL; k0 += TILE_K) {
    // Stage A tile: 128 x 32 fp32 -> bf16 (1024 float4 chunks / 256 thr)
#pragma unroll
    for (int it = 0; it < 4; ++it) {
      int i = tid + it * 256;
      int r = i >> 3, c = (i & 7) << 2;
      float4 v = *(const float4*)&x[(size_t)(m0 + r) * D_MODEL + k0 + c];
      bf16* p = &As[r * LDS_A + c];
      p[0] = (bf16)v.x; p[1] = (bf16)v.y; p[2] = (bf16)v.z; p[3] = (bf16)v.w;
    }
    // Stage B tile: 64 x 32 fp32 -> bf16 (512 float4 chunks)
#pragma unroll
    for (int it = 0; it < 2; ++it) {
      int i = tid + it * 256;
      int r = i >> 3, c = (i & 7) << 2;
      float4 v = *(const float4*)&W[(size_t)(n0 + r) * D_MODEL + k0 + c];
      bf16* p = &Bs[r * LDS_B + c];
      p[0] = (bf16)v.x; p[1] = (bf16)v.y; p[2] = (bf16)v.z; p[3] = (bf16)v.w;
    }
    __syncthreads();

    v16bf af0 = frag_a(&As[(wm * 32 + 0)  * LDS_A], LDS_A, lane);
    v16bf af1 = frag_a(&As[(wm * 32 + 16) * LDS_A], LDS_A, lane);
    v16bf bf0 = frag_b(&Bs[(wn * 32 + 0)  * LDS_B], LDS_B, lane);
    v16bf bf1 = frag_b(&Bs[(wn * 32 + 16) * LDS_B], LDS_B, lane);
    acc[0][0] = wmma_bf16(af0, bf0, acc[0][0]);
    acc[0][1] = wmma_bf16(af0, bf1, acc[0][1]);
    acc[1][0] = wmma_bf16(af1, bf0, acc[1][0]);
    acc[1][1] = wmma_bf16(af1, bf1, acc[1][1]);
    __syncthreads();
  }

  // Epilogue: add bias, scale, scatter to per-head layout
  const int n  = lane & 15;
  const int hi = lane >> 4;
#pragma unroll
  for (int i = 0; i < 2; ++i)
#pragma unroll
    for (int j = 0; j < 2; ++j) {
      int col = n0 + wn * 32 + j * 16 + n;
      float bval = bias[col];
      int hh = col >> 6, d = col & 63;
#pragma unroll
      for (int r = 0; r < 8; ++r) {
        int row = m0 + wm * 32 + i * 16 + r + hi * 8;
        float v = (acc[i][j][r] + bval) * oscale;
        int bb = row >> 11, s = row & (SEQ - 1);
        size_t headbase = (size_t)(bb * HEADS + hh);
        if (z == 2) {
          // V transposed: [B][H][DK][S]
          out[(headbase * DK + d) * SEQ + s] = (bf16)v;
        } else {
          // Q/K: [B][H][S][DK]
          out[(headbase * SEQ + s) * DK + d] = (bf16)v;
        }
      }
    }
}

// Issue the async copies for one 64-key K/V tile pair into the given buffers.
__device__ __forceinline__ void issue_kv_tile(const bf16* __restrict__ Kh,
                                              const bf16* __restrict__ Vh,
                                              int k0, int tid,
                                              bf16* ksb, bf16* vsb) {
#pragma unroll
  for (int it = 0; it < 2; ++it) {
    int i = tid + it * 256;
    int r = i >> 3, c = (i & 7) << 3;
    copy16_g2l(Kh + (size_t)(k0 + r) * DK + c, &ksb[r * ATT_LD + c]);
    copy16_g2l(Vh + (size_t)r * SEQ + k0 + c, &vsb[r * ATT_LD + c]);
  }
}

// ---------------------------------------------------------------------------
// Kernel 2: causal flash attention.  Q/K: [B][H][S][DK], V: [B][H][DK][S],
// all bf16.  Block = 128 q-rows (8 waves x 16 rows); 64-key K/V tiles are
// DOUBLE-BUFFERED: tile t+1 streams into LDS via GLOBAL_LOAD_ASYNC_TO_LDS_B128
// while tile t is consumed by the WMMAs; S_WAIT_ASYNCCNT gates consumption.
// Q pre-scaled so softmax uses exp2 directly.  Output -> [B][S][D] bf16.
// grid = (S/128, H, B), block = 256
// ---------------------------------------------------------------------------
__global__ __launch_bounds__(256)
void flash_attn_kernel(const bf16* __restrict__ Q, const bf16* __restrict__ K,
                       const bf16* __restrict__ V, bf16* __restrict__ O) {
  __shared__ bf16 Ks[2][64 * ATT_LD];       // [buf][key][d]
  __shared__ bf16 Vs[2][64 * ATT_LD];       // [buf][d][key] (V pre-transposed)
  __shared__ bf16 Ps[8 * 16 * ATT_LD];      // per-wave P scratch

  const int qt = blockIdx.x;
  const int h  = blockIdx.y;
  const int b  = blockIdx.z;
  const int tid  = threadIdx.x;
  const int wave = tid >> 5;
  const int lane = tid & 31;
  const int n  = lane & 15;
  const int hi = lane >> 4;

  const size_t head = (size_t)(b * HEADS + h);
  const bf16* Qh = Q + head * SEQ * DK;
  const bf16* Kh = K + head * SEQ * DK;
  const bf16* Vh = V + head * DK * SEQ;     // transposed layout

  const int q0 = qt * 128 + wave * 16;      // this wave's 16 query rows

  // Load Q fragments once (A-matrix layout, ld = DK)
  v16bf qf0 = frag_a(Qh + (size_t)q0 * DK,      DK, lane);
  v16bf qf1 = frag_a(Qh + (size_t)q0 * DK + 32, DK, lane);

  v8f oacc[4];
#pragma unroll
  for (int of = 0; of < 4; ++of) oacc[of] = v8f_zero();
  float mrow[8], lrow[8];
#pragma unroll
  for (int r = 0; r < 8; ++r) { mrow[r] = -1e30f; lrow[r] = 0.f; }

  const int kend = qt * 128 + 128;          // causal bound for this block

  // Prologue: start the async stream for tile 0.
  issue_kv_tile(Kh, Vh, 0, tid, Ks[0], Vs[0]);

  int t = 0;
  for (int k0 = 0; k0 < kend; k0 += 64, t ^= 1) {
    // Tile t has landed when our ASYNCcnt drains; barrier makes it visible
    // block-wide and guarantees buf t^1 is no longer being read.
    async_copy_fence();
    __syncthreads();

    // Kick off tile t+1 into the other buffer; it streams during compute.
    if (k0 + 64 < kend)
      issue_kv_tile(Kh, Vh, k0 + 64, tid, Ks[t ^ 1], Vs[t ^ 1]);
    // L2 prefetch two tiles ahead.
    if (k0 + 128 < kend) {
      __builtin_prefetch(Kh + (size_t)(k0 + 128) * DK + tid * 16, 0, 1);
      __builtin_prefetch(Vh + (size_t)(tid >> 2) * SEQ + (k0 + 128) + (tid & 3) * 16, 0, 1);
    }

    const bf16* ksb = Ks[t];
    const bf16* vsb = Vs[t];

    if (k0 <= q0 + 15) {                    // wave has unmasked work
      // ---- scores: S = Q K^T, 16 x 64, fp32 acc ----
      v8f sfr[4];
#pragma unroll
      for (int j = 0; j < 4; ++j) {
        v8f s = v8f_zero();
        s = wmma_bf16(qf0, frag_b(&ksb[(j * 16) * ATT_LD + 0],  ATT_LD, lane), s);
        s = wmma_bf16(qf1, frag_b(&ksb[(j * 16) * ATT_LD + 32], ATT_LD, lane), s);
        sfr[j] = s;
      }
      // ---- causal mask ----
#pragma unroll
      for (int j = 0; j < 4; ++j) {
        int key = k0 + j * 16 + n;
#pragma unroll
        for (int r = 0; r < 8; ++r) {
          int q = q0 + r + hi * 8;
          if (key > q) sfr[j][r] = -1e30f;
        }
      }
      // ---- row max (within 16-lane group) ----
      float mnew[8];
#pragma unroll
      for (int r = 0; r < 8; ++r)
        mnew[r] = fmaxf(fmaxf(sfr[0][r], sfr[1][r]),
                        fmaxf(sfr[2][r], sfr[3][r]));
#pragma unroll
      for (int msk = 1; msk < 16; msk <<= 1)
#pragma unroll
        for (int r = 0; r < 8; ++r)
          mnew[r] = fmaxf(mnew[r], __shfl_xor(mnew[r], msk, 32));
      float alpha[8];
#pragma unroll
      for (int r = 0; r < 8; ++r) {
        mnew[r]  = fmaxf(mnew[r], mrow[r]);
        alpha[r] = exp2f(mrow[r] - mnew[r]);
        mrow[r]  = mnew[r];
      }
      // ---- P = exp2(S - m), row sums ----
      float rsum[8];
#pragma unroll
      for (int r = 0; r < 8; ++r) rsum[r] = 0.f;
#pragma unroll
      for (int j = 0; j < 4; ++j)
#pragma unroll
        for (int r = 0; r < 8; ++r) {
          float p = exp2f(sfr[j][r] - mnew[r]);
          sfr[j][r] = p;
          rsum[r] += p;
        }
#pragma unroll
      for (int msk = 1; msk < 16; msk <<= 1)
#pragma unroll
        for (int r = 0; r < 8; ++r)
          rsum[r] += __shfl_xor(rsum[r], msk, 32);
#pragma unroll
      for (int r = 0; r < 8; ++r)
        lrow[r] = lrow[r] * alpha[r] + rsum[r];
      // ---- rescale O accumulators ----
#pragma unroll
      for (int of = 0; of < 4; ++of)
#pragma unroll
        for (int r = 0; r < 8; ++r) oacc[of][r] *= alpha[r];

      // ---- stage P (C-layout -> A-layout) through wave-private LDS ----
      bf16* pw = &Ps[wave * 16 * ATT_LD];
#pragma unroll
      for (int j = 0; j < 4; ++j)
#pragma unroll
        for (int r = 0; r < 8; ++r)
          pw[(r + hi * 8) * ATT_LD + j * 16 + n] = (bf16)sfr[j][r];

      v16bf pf0 = frag_a(pw,      ATT_LD, lane);   // keys 0..31
      v16bf pf1 = frag_a(pw + 32, ATT_LD, lane);   // keys 32..63

      // ---- O += P @ V ----
#pragma unroll
      for (int of = 0; of < 4; ++of) {
        oacc[of] = wmma_bf16(pf0, frag_b(&vsb[(of * 16) * ATT_LD + 0],  ATT_LD, lane), oacc[of]);
        oacc[of] = wmma_bf16(pf1, frag_b(&vsb[(of * 16) * ATT_LD + 32], ATT_LD, lane), oacc[of]);
      }
    }
    // No trailing barrier: the fence+barrier at the top of the next
    // iteration orders buffer reuse.
  }

  // ---- normalize + write O as [B][S][D] bf16 ----
  float inv[8];
#pragma unroll
  for (int r = 0; r < 8; ++r) inv[r] = 1.0f / lrow[r];
#pragma unroll
  for (int of = 0; of < 4; ++of) {
    int d = h * DK + of * 16 + n;
#pragma unroll
    for (int r = 0; r < 8; ++r) {
      int q = q0 + r + hi * 8;
      O[((size_t)b * SEQ + q) * D_MODEL + d] = (bf16)(oacc[of][r] * inv[r]);
    }
  }
}

// ---------------------------------------------------------------------------
// Kernel 3: output projection.  out = O @ Wo^T + bo  (bf16 A, fp32 W/out)
// A tile streamed with async b128 copies (pure bf16 copy, no conversion).
// grid = (M/128, N/64), block = 256
// ---------------------------------------------------------------------------
__global__ __launch_bounds__(256)
void out_proj_kernel(const bf16* __restrict__ A, const float* __restrict__ W,
                     const float* __restrict__ bias, float* __restrict__ out) {
  __shared__ bf16 As[TILE_M * LDS_A];
  __shared__ bf16 Bs[TILE_N * LDS_B];

  const int tid  = threadIdx.x;
  const int wave = tid >> 5;
  const int lane = tid & 31;
  const int wm = wave & 3;
  const int wn = wave >> 2;
  const int m0 = blockIdx.x * TILE_M;
  const int n0 = blockIdx.y * TILE_N;

  v8f acc[2][2];
#pragma unroll
  for (int i = 0; i < 2; ++i)
#pragma unroll
    for (int j = 0; j < 2; ++j) acc[i][j] = v8f_zero();

  for (int k0 = 0; k0 < D_MODEL; k0 += TILE_K) {
    // A tile already bf16: 128 x 32 -> 512 b128 chunks, async copy to LDS
#pragma unroll
    for (int it = 0; it < 2; ++it) {
      int i = tid + it * 256;
      int r = i >> 2, c = (i & 3) << 3;
      copy16_g2l(A + (size_t)(m0 + r) * D_MODEL + k0 + c, &As[r * LDS_A + c]);
    }
    // B tile fp32 -> bf16 (conversion, stays on the VALU path)
#pragma unroll
    for (int it = 0; it < 2; ++it) {
      int i = tid + it * 256;
      int r = i >> 3, c = (i & 7) << 2;
      float4 v = *(const float4*)&W[(size_t)(n0 + r) * D_MODEL + k0 + c];
      bf16* p = &Bs[r * LDS_B + c];
      p[0] = (bf16)v.x; p[1] = (bf16)v.y; p[2] = (bf16)v.z; p[3] = (bf16)v.w;
    }
    async_copy_fence();
    __syncthreads();

    v16bf af0 = frag_a(&As[(wm * 32 + 0)  * LDS_A], LDS_A, lane);
    v16bf af1 = frag_a(&As[(wm * 32 + 16) * LDS_A], LDS_A, lane);
    v16bf bf0 = frag_b(&Bs[(wn * 32 + 0)  * LDS_B], LDS_B, lane);
    v16bf bf1 = frag_b(&Bs[(wn * 32 + 16) * LDS_B], LDS_B, lane);
    acc[0][0] = wmma_bf16(af0, bf0, acc[0][0]);
    acc[0][1] = wmma_bf16(af0, bf1, acc[0][1]);
    acc[1][0] = wmma_bf16(af1, bf0, acc[1][0]);
    acc[1][1] = wmma_bf16(af1, bf1, acc[1][1]);
    __syncthreads();
  }

  const int n  = lane & 15;
  const int hi = lane >> 4;
#pragma unroll
  for (int i = 0; i < 2; ++i)
#pragma unroll
    for (int j = 0; j < 2; ++j) {
      int col = n0 + wn * 32 + j * 16 + n;
      float bval = bias[col];
#pragma unroll
      for (int r = 0; r < 8; ++r) {
        int row = m0 + wm * 32 + i * 16 + r + hi * 8;
        out[(size_t)row * D_MODEL + col] = acc[i][j][r] + bval;
      }
    }
}

// ---------------------------------------------------------------------------
extern "C" void kernel_launch(void* const* d_in, const int* in_sizes, int n_in,
                              void* d_out, int out_size, void* d_ws,
                              size_t ws_size, hipStream_t stream) {
  const float* x  = (const float*)d_in[0];
  const float* Wq = (const float*)d_in[1];
  const float* bq = (const float*)d_in[2];
  const float* Wk = (const float*)d_in[3];
  const float* bk = (const float*)d_in[4];
  const float* Wv = (const float*)d_in[5];
  const float* bv = (const float*)d_in[6];
  const float* Wo = (const float*)d_in[7];
  const float* bo = (const float*)d_in[8];
  float* out = (float*)d_out;

  const size_t elems = (size_t)M_TOT * D_MODEL;  // 4096*1024
  bf16* Qw = (bf16*)d_ws;
  bf16* Kw = Qw + elems;
  bf16* Vw = Kw + elems;
  bf16* Ow = Vw + elems;

  dim3 blk(256);
  dim3 g1(M_TOT / TILE_M, D_MODEL / TILE_N, 3);        // 32 x 16 x 3
  qkv_proj_kernel<<<g1, blk, 0, stream>>>(x, Wq, Wk, Wv, bq, bk, bv,
                                          Qw, Kw, Vw);

  dim3 g2(SEQ / 128, HEADS, BATCH);                    // 16 x 16 x 2
  flash_attn_kernel<<<g2, blk, 0, stream>>>(Qw, Kw, Vw, Ow);

  dim3 g3(M_TOT / TILE_M, D_MODEL / TILE_N);           // 32 x 16
  out_proj_kernel<<<g3, blk, 0, stream>>>(Ow, Wo, bo, out);
}
